// LSTMcell_17480516895199
// MI455X (gfx1250) — compile-verified, matrix operand order checked by reading, not run on hardware
//
#include <hip/hip_runtime.h>
#include <math.h>

// ---- CDNA5 (gfx1250) vector types for WMMA ----
typedef __attribute__((ext_vector_type(16))) __bf16 v16bf;
typedef __attribute__((ext_vector_type(8)))  __bf16 v8bf;
typedef __attribute__((ext_vector_type(4)))  __bf16 v4bf;
typedef __attribute__((ext_vector_type(8)))  float  v8f;

#define B_DIM  4096
#define IN_DIM 2048
#define H_DIM  2048

#define BLOCK 512     // 16 waves
#define BM 256        // batch rows per workgroup (16 waves x 16 rows)
#define BN 16         // hidden units per workgroup (x4 gates)
#define BK 32         // K-slice == WMMA bf16 K
#define PITCH 40      // bf16 elems per LDS row: 32 data + 8 pad (80B, conflict-free b128 reads)
#define NT  ((IN_DIM + H_DIM) / BK)   // 128 K-tiles total (phase 0: x/Wx, phase 1: h/Wh)
#define NPH (IN_DIM / BK)             // 64 tiles per phase

static __device__ __forceinline__ float sigmoid_f(float v) {
  return 1.0f / (1.0f + __expf(-v));
}
static __device__ __forceinline__ float tanh_f(float v) {
  float e = __expf(-2.0f * v);
  return (1.0f - e) / (1.0f + e);
}

__global__ __launch_bounds__(BLOCK)
void lstm_cell_wmma(const float* __restrict__ x,   const float* __restrict__ h,
                    const float* __restrict__ c,
                    const float* __restrict__ Wii, const float* __restrict__ bii,
                    const float* __restrict__ Wif, const float* __restrict__ bif,
                    const float* __restrict__ Wig, const float* __restrict__ big,
                    const float* __restrict__ Wio, const float* __restrict__ bio,
                    const float* __restrict__ Whi, const float* __restrict__ bhi,
                    const float* __restrict__ Whf, const float* __restrict__ bhf,
                    const float* __restrict__ Whg, const float* __restrict__ bhg,
                    const float* __restrict__ Who, const float* __restrict__ bho,
                    float* __restrict__ out)
{
  // double-buffered staging tiles (bf16): A 2x20KB, W 2x5KB = 50 KB
  __shared__ __align__(16) __bf16 sA[2][BM * PITCH];
  __shared__ __align__(16) __bf16 sW[2][4 * BN * PITCH];

  const int t    = threadIdx.x;
  const int lane = t & 31;
  const int wave = t >> 5;                 // 0..15 -> 16-row M slab
  const int m0   = blockIdx.y * BM;
  const int n0   = blockIdx.x * BN;

  // staging coords: each thread moves 4 float4 of A and 1 float4 of W per tile
  const int ldRow = t >> 3;                // 0..63  (A rows, +i*64)
  const int ldCol = (t & 7) * 4;           // 0,4,..,28
  const int wGate = t >> 7;                // 0..3, wave-uniform gate select
  const int wRow  = (t >> 3) & 15;         // 0..15  (W rows)

  // thread-varying parts of global addresses (32-bit, shared across loads)
  const int aOffT = ldRow * IN_DIM + ldCol;
  const int wOffT = wRow  * IN_DIM + ldCol;

  // per-phase weight pointers for this thread's gate (loop-invariant)
  const float* wpx = (wGate == 0 ? Wii : wGate == 1 ? Wif : wGate == 2 ? Wig : Wio)
                     + n0 * IN_DIM + wOffT;
  const float* wph = (wGate == 0 ? Whi : wGate == 1 ? Whf : wGate == 2 ? Whg : Who)
                     + n0 * IN_DIM + wOffT;

  // WMMA fragment coords (16-bit A/B VGPR layout, wave32)
  const int fl   = lane & 15;
  const int fh   = lane >> 4;              // K-half select
  const int koff = fh * 8;                 // element offset of first K chunk

  v8f acc[4];                              // [gate]
  {
    v8f z = {};
#pragma unroll
    for (int g = 0; g < 4; ++g) acc[g] = z;
  }

  float4 ra[4], rw;

  // issue global loads (f32) for K-tile `it`
  auto loadTile = [&](int it) {
    const int   kg = (it & (NPH - 1)) * BK;
    const bool  p1 = it >= NPH;
    const float* aB = (p1 ? h : x) + m0 * IN_DIM + kg;     // block-uniform base
#pragma unroll
    for (int i = 0; i < 4; ++i)
      ra[i] = *(const float4*)(aB + aOffT + i * (64 * IN_DIM));
    rw = *(const float4*)((p1 ? wph : wpx) + kg);
  };

  // convert f32 -> bf16 and store staged tile into LDS buffer `d` (compile-time d)
  auto storeTile = [&](int d) {
#pragma unroll
    for (int i = 0; i < 4; ++i) {
      v4bf pk;
      pk[0] = (__bf16)ra[i].x; pk[1] = (__bf16)ra[i].y;
      pk[2] = (__bf16)ra[i].z; pk[3] = (__bf16)ra[i].w;
      *(v4bf*)&sA[d][(i * 64 + ldRow) * PITCH + ldCol] = pk;
    }
    {
      v4bf pk;
      pk[0] = (__bf16)rw.x; pk[1] = (__bf16)rw.y;
      pk[2] = (__bf16)rw.z; pk[3] = (__bf16)rw.w;
      *(v4bf*)&sW[d][(wGate * BN + wRow) * PITCH + ldCol] = pk;
    }
  };

  // compute from buffer `d` (compile-time d): 1 A fragment, 4 B fragments, 4 WMMAs
  auto computeTile = [&](int d) {
    const __bf16* As = &sA[d][0];
    const __bf16* Ws = &sW[d][0];

    const int arow = (wave * 16 + fl) * PITCH + koff;
    v8bf a0 = *(const v8bf*)&As[arow];
    v8bf a1 = *(const v8bf*)&As[arow + 16];
    v16bf af;
#pragma unroll
    for (int q = 0; q < 8; ++q) { af[q] = a0[q]; af[8 + q] = a1[q]; }

    v16bf bv[4];
#pragma unroll
    for (int g = 0; g < 4; ++g) {
      const int brow = (g * BN + fl) * PITCH + koff;
      v8bf b0 = *(const v8bf*)&Ws[brow];
      v8bf b1 = *(const v8bf*)&Ws[brow + 16];
#pragma unroll
      for (int e = 0; e < 8; ++e) { bv[g][e] = b0[e]; bv[g][8 + e] = b1[e]; }
    }
#pragma unroll
    for (int g = 0; g < 4; ++g)
      acc[g] = __builtin_amdgcn_wmma_f32_16x16x32_bf16(
          false, af, false, bv[g], (short)0, acc[g], false, false);
  };

  // ---- pipeline prologue: tile 0 -> buffer 0 ----
  loadTile(0);
  storeTile(0);
  __syncthreads();

  // ---- main loop, manually unrolled 2x so buffer indices are compile-time ----
#pragma unroll 1
  for (int it = 0; it < NT; it += 2) {
    // half 0: compute buf0 (tile it), stage tile it+1 -> buf1
    loadTile(it + 1);                      // it+1 <= NT-1 always (NT even)
    computeTile(0);
    storeTile(1);
    __syncthreads();

    // half 1: compute buf1 (tile it+1), stage tile it+2 -> buf0
    const bool more = (it + 2 < NT);
    if (more) loadTile(it + 2);
    computeTile(1);
    if (more) {
      storeTile(0);
      __syncthreads();
    }
  }

  // ---- fused LSTM epilogue: biases + nonlinearities + state update ----
  const int col = n0 + fl;
  const float bi2 = bii[col] + bhi[col];
  const float bf2 = bif[col] + bhf[col];
  const float bg2 = big[col] + bhg[col];
  const float bo2 = bio[col] + bho[col];
#pragma unroll
  for (int e = 0; e < 8; ++e) {
    const int row = m0 + wave * 16 + fh * 8 + e;   // C-matrix VGPR layout
    const float iv = sigmoid_f(acc[0][e] + bi2);
    const float fv = sigmoid_f(acc[1][e] + bf2);
    const float gv = tanh_f  (acc[2][e] + bg2);
    const float ov = sigmoid_f(acc[3][e] + bo2);
    const float cold = c[row * H_DIM + col];
    const float cn = fv * cold + iv * gv;
    const float hn = ov * tanh_f(cn);
    out[row * H_DIM + col] = hn;                       // h_new
    out[B_DIM * H_DIM + row * H_DIM + col] = cn;       // c_new
  }
}

extern "C" void kernel_launch(void* const* d_in, const int* in_sizes, int n_in,
                              void* d_out, int out_size, void* d_ws, size_t ws_size,
                              hipStream_t stream) {
  (void)in_sizes; (void)n_in; (void)out_size; (void)d_ws; (void)ws_size;
  const float* x   = (const float*)d_in[0];
  const float* h   = (const float*)d_in[1];
  const float* c   = (const float*)d_in[2];
  const float* Wii = (const float*)d_in[3];   const float* bii = (const float*)d_in[4];
  const float* Wif = (const float*)d_in[5];   const float* bif = (const float*)d_in[6];
  const float* Wig = (const float*)d_in[7];   const float* big = (const float*)d_in[8];
  const float* Wio = (const float*)d_in[9];   const float* bio = (const float*)d_in[10];
  const float* Whi = (const float*)d_in[11];  const float* bhi = (const float*)d_in[12];
  const float* Whf = (const float*)d_in[13];  const float* bhf = (const float*)d_in[14];
  const float* Whg = (const float*)d_in[15];  const float* bhg = (const float*)d_in[16];
  const float* Who = (const float*)d_in[17];  const float* bho = (const float*)d_in[18];

  dim3 grid(H_DIM / BN, B_DIM / BM);   // (128, 16)
  dim3 block(BLOCK);
  lstm_cell_wmma<<<grid, block, 0, stream>>>(
      x, h, c,
      Wii, bii, Wif, bif, Wig, big, Wio, bio,
      Whi, bhi, Whf, bhf, Whg, bhg, Who, bho,
      (float*)d_out);
}
